// Attention_18056042513007
// MI455X (gfx1250) — matrix operand order repaired and checked by reading, not compiled, not run on hardware
//
#include <hip/hip_runtime.h>
#include <hip/hip_bf16.h>
#include <stdint.h>

typedef __bf16 bf16;
typedef __attribute__((ext_vector_type(16))) __bf16 v16bf;
typedef __attribute__((ext_vector_type(8)))  __bf16 v8bf;
typedef __attribute__((ext_vector_type(8)))  float  v8f;

static constexpr int B_ = 2, T_ = 2048, D_ = 1024, H_ = 16, DH_ = 64;
static constexpr float SCORE_NEG = -1e30f;

#define WMMA_BF16(A, Bm, C) \
    __builtin_amdgcn_wmma_f32_16x16x32_bf16(false, (A), false, (Bm), (short)0, (C), false, false)

__device__ __forceinline__ v8f vzero() {
    v8f z;
#pragma unroll
    for (int i = 0; i < 8; ++i) z[i] = 0.0f;
    return z;
}

// ---- deterministic pseudo-gaussian noise (stands in for jax key(42) noise) ----
__device__ __forceinline__ float noisef(int bh, int q, int t) {
    uint32_t x = (uint32_t)((bh * T_ + q) * T_ + t);
    x *= 2654435761u; x ^= x >> 17;
    x *= 0xed5ad4bbu; x ^= x >> 11;
    x *= 0xac4c1b51u; x ^= x >> 15;
    float u1 = (float)(x & 0xffffu) * (1.0f / 65536.0f);
    float u2 = (float)(x >> 16)     * (1.0f / 65536.0f);
    return 0.01f * 2.449489743f * (u1 + u2 - 1.0f);
}

// =====================================================================
// Kernel 0a: fp32 -> bf16 elementwise convert (residual stream)
// =====================================================================
__global__ void cvt_bf16_kernel(const float* __restrict__ src, bf16* __restrict__ dst, int n) {
    int i = blockIdx.x * blockDim.x + threadIdx.x;
    if (i < n) dst[i] = (bf16)src[i];
}

// =====================================================================
// Kernel 0b: pack row-major [K][N] fp32 matrices into CDNA5 B-matrix
// bf16 chunks: chunk(kc,nc) holds K in [32kc,32kc+32), N in [16nc,16nc+16);
// within a chunk: offset = lane*16 + e with lane = (n%16) + 16*((k%16)/8),
// e = k%8 + 8*((k%32)/16).  One contiguous v16bf per lane at WMMA time.
// =====================================================================
__global__ void pack_b_kernel(const float* __restrict__ src, bf16* __restrict__ dst,
                              int K, int N, int nmats) {
    int per = K * N;
    int idx = blockIdx.x * blockDim.x + threadIdx.x;
    if (idx >= per * nmats) return;
    int m = idx / per, o = idx % per;
    int chunk = o >> 9, r = o & 511;
    int lane = r >> 4, e = r & 15;
    int ncch = N >> 4;
    int kc = chunk / ncch, nc = chunk % ncch;
    int k = kc * 32 + (e >> 3) * 16 + (lane >> 4) * 8 + (e & 7);
    int n = nc * 16 + (lane & 15);
    dst[idx] = (bf16)src[m * per + k * N + n];
}

// =====================================================================
// Kernel 1: QKV projection.  One wave computes a 16(T) x 64(Dh) tile of
// Q, K and V for one (b,h).  12 WMMAs per K-step, 32 K-steps = 384 WMMAs.
// Explicitly pipelined: next chunk's operands are loaded, then a scheduling
// fence (memory clobber + accumulators tied) keeps the loads above and the
// WMMAs below, so waits become partial instead of s_wait_loadcnt 0.
// =====================================================================
__global__ __launch_bounds__(32) void qkv_kernel(
    const bf16* __restrict__ res, const bf16* __restrict__ wq,
    const bf16* __restrict__ wk, const bf16* __restrict__ wv,
    const float* __restrict__ bq, const float* __restrict__ bk,
    const float* __restrict__ bv,
    bf16* __restrict__ qout, bf16* __restrict__ kpack, bf16* __restrict__ vpack) {
    int bid = blockIdx.x;
    int qt = bid & 127;          // T/16 tiles
    int bh = bid >> 7;
    int b  = bh >> 4;
    int h  = bh & 15;
    int lane = threadIdx.x, lm = lane & 15, lg = lane >> 4;
    int t0 = qt * 16;

    v8f accq[4], acck[4], accv[4];
#pragma unroll
    for (int i = 0; i < 4; ++i) { accq[i] = vzero(); acck[i] = vzero(); accv[i] = vzero(); }

    const bf16* resrow = res + (b * T_ + t0 + lm) * D_;   // A row = t0+lm
    const int wstride = (D_ / 32) * (DH_ / 16) * 512;     // 65536 per head
    const bf16* wqh = wq + h * wstride;
    const bf16* wkh = wk + h * wstride;
    const bf16* wvh = wv + h * wstride;

    // ---- prologue: A(0) and B(chunk 0) ----
    v8bf a0 = *(const v8bf*)(resrow + lg * 8);
    v8bf a1 = *(const v8bf*)(resrow + 16 + lg * 8);
    int c0 = lane * 16;
    v16bf cbq = *(const v16bf*)(wqh + c0);
    v16bf cbk = *(const v16bf*)(wkh + c0);
    v16bf cbv = *(const v16bf*)(wvh + c0);

    for (int kc = 0; kc < D_ / 32; ++kc) {
        v16bf a;
#pragma unroll
        for (int i = 0; i < 8; ++i) { a[i] = a0[i]; a[8 + i] = a1[i]; }
        // preload next A tile while this K-step computes (tail overrun stays
        // inside the workspace and is never consumed)
        a0 = *(const v8bf*)(resrow + (kc + 1) * 32 + lg * 8);
        a1 = *(const v8bf*)(resrow + (kc + 1) * 32 + 16 + lg * 8);
#pragma unroll
        for (int nt = 0; nt < 4; ++nt) {
            // issue next chunk's B loads
            int noff = (kc * 4 + nt + 1) * 512 + lane * 16;
            v16bf nbq = *(const v16bf*)(wqh + noff);
            v16bf nbk = *(const v16bf*)(wkh + noff);
            v16bf nbv = *(const v16bf*)(wvh + noff);
            // scheduling fence: loads stay above, WMMAs stay below
            asm volatile("" : "+v"(accq[nt]), "+v"(acck[nt]), "+v"(accv[nt]) :: "memory");
            accq[nt] = WMMA_BF16(a, cbq, accq[nt]);
            acck[nt] = WMMA_BF16(a, cbk, acck[nt]);
            accv[nt] = WMMA_BF16(a, cbv, accv[nt]);
            cbq = nbq; cbk = nbk; cbv = nbv;
        }
    }

    bf16* qbase = qout + (bh * T_ + t0) * DH_;
    int half = qt & 1;  // which 16-half of the 32-wide t chunk
#pragma unroll
    for (int nt = 0; nt < 4; ++nt) {
        int d = nt * 16 + lm;
        float bqv = bq[h * DH_ + d], bkv = bk[h * DH_ + d], bvv = bv[h * DH_ + d];
        bf16* kch = kpack + (bh * 2 * (T_ / 16) + (nt >> 1) * (T_ / 16) + qt) * 512;
        bf16* vch = vpack + (bh * (T_ / 32) * 4 + (t0 >> 5) * 4 + nt) * 512;
        v8bf vvec;
#pragma unroll
        for (int r = 0; r < 8; ++r) {
            int m = r + 8 * lg;                       // t offset within tile
            qbase[m * DH_ + d] = (bf16)(accq[nt][r] + bqv);
            // K B-layout: lane2 = (t%16) + 16*((d%16)/8), e = d%8 + 8*((d%32)/16)
            kch[(m + 16 * (lm >> 3)) * 16 + 8 * (nt & 1) + (lm & 7)] = (bf16)(acck[nt][r] + bkv);
            vvec[r] = (bf16)(accv[nt][r] + bvv);
        }
        // V B-layout: lane2 = (d%16) + 16*((t%16)/8), e = t%8 + 8*((t%32)/16)
        *(v8bf*)(vch + (lm + 16 * lg) * 16 + 8 * half) = vvec;
    }
}

// =====================================================================
// Kernel 2: flash attention per (b, h, 16-row q tile).  Online softmax,
// causal mask + noise, P transposed C->A layout through LDS.  V operand
// loads are fenced above the softmax VALU section so they retire under
// the exp/shuffle work.
// =====================================================================
__global__ __launch_bounds__(32) void attn_kernel(
    const bf16* __restrict__ qbuf, const bf16* __restrict__ kpack,
    const bf16* __restrict__ vpack, bf16* __restrict__ zout) {
    __shared__ __align__(32) bf16 ptile[16 * 32];

    int bid = blockIdx.x;
    int qt = bid & 127;
    int bh = bid >> 7;
    int b  = bh >> 4;
    int h  = bh & 15;
    int lane = threadIdx.x, lm = lane & 15, lg = lane >> 4;
    int t0 = qt * 16;

    // Q as A-matrix, K-dim = Dh = 64 -> two A registers
    const bf16* qrow = qbuf + (bh * T_ + t0 + lm) * DH_;
    v16bf aQ[2];
#pragma unroll
    for (int ac = 0; ac < 2; ++ac) {
        v8bf x0 = *(const v8bf*)(qrow + ac * 32 + lg * 8);
        v8bf x1 = *(const v8bf*)(qrow + ac * 32 + 16 + lg * 8);
#pragma unroll
        for (int i = 0; i < 8; ++i) { aQ[ac][i] = x0[i]; aQ[ac][8 + i] = x1[i]; }
    }

    const bf16* kbh = kpack + bh * 2 * (T_ / 16) * 512;
    const bf16* vbh = vpack + bh * (T_ / 32) * 4 * 512;

    float mrun[8], lrun[8];
    v8f zacc[4];
#pragma unroll
    for (int r = 0; r < 8; ++r) { mrun[r] = SCORE_NEG; lrun[r] = 0.0f; }
#pragma unroll
    for (int nt = 0; nt < 4; ++nt) zacc[nt] = vzero();

    int nchunk = (t0 + 15) / 32 + 1;   // causal: only k-chunks reaching t <= q
    for (int tc = 0; tc < nchunk; ++tc) {
        // ---- scores: two 16x16 tiles, K = 64 over two dc chunks ----
        v8f s[2];
#pragma unroll
        for (int n2 = 0; n2 < 2; ++n2) {
            int tt = tc * 2 + n2;
            v8f c = vzero();
            c = WMMA_BF16(aQ[0], *(const v16bf*)(kbh + tt * 512 + lane * 16), c);
            c = WMMA_BF16(aQ[1], *(const v16bf*)(kbh + ((T_ / 16) + tt) * 512 + lane * 16), c);
            s[n2] = c;
        }
        // ---- issue V operand loads; fence keeps them above the softmax ----
        v16bf vb[4];
#pragma unroll
        for (int nt = 0; nt < 4; ++nt)
            vb[nt] = *(const v16bf*)(vbh + (tc * 4 + nt) * 512 + lane * 16);
        asm volatile("" : "+v"(s[0]), "+v"(s[1]) :: "memory");

        // ---- mask + noise + online softmax (rows across 16-lane halves) ----
#pragma unroll
        for (int r = 0; r < 8; ++r) {
            int qg = t0 + r + 8 * lg;
            int tg0 = tc * 32 + lm;
            int tg1 = tc * 32 + 16 + lm;
            float s0 = (tg0 <= qg) ? (s[0][r] * 0.125f + noisef(bh, qg, tg0)) : SCORE_NEG;
            float s1 = (tg1 <= qg) ? (s[1][r] * 0.125f + noisef(bh, qg, tg1)) : SCORE_NEG;
            float mx = fmaxf(s0, s1);
#pragma unroll
            for (int o = 1; o < 16; o <<= 1) mx = fmaxf(mx, __shfl_xor(mx, o, 32));
            float mnew = fmaxf(mrun[r], mx);
            float p0 = __expf(s0 - mnew);
            float p1 = __expf(s1 - mnew);
            float rs = p0 + p1;
#pragma unroll
            for (int o = 1; o < 16; o <<= 1) rs += __shfl_xor(rs, o, 32);
            float sc = __expf(mrun[r] - mnew);
            lrun[r] = lrun[r] * sc + rs;
            mrun[r] = mnew;
#pragma unroll
            for (int nt = 0; nt < 4; ++nt) zacc[nt][r] *= sc;
            int m = r + 8 * lg;
            ptile[m * 32 + lm]      = (bf16)p0;
            ptile[m * 32 + 16 + lm] = (bf16)p1;
        }
        __syncthreads();
        // ---- read P back in A-layout (two 8-element runs per lane) ----
        v16bf aP;
        {
            v8bf x0 = *(const v8bf*)(&ptile[lm * 32 + lg * 8]);
            v8bf x1 = *(const v8bf*)(&ptile[lm * 32 + 16 + lg * 8]);
#pragma unroll
            for (int i = 0; i < 8; ++i) { aP[i] = x0[i]; aP[8 + i] = x1[i]; }
        }
#pragma unroll
        for (int nt = 0; nt < 4; ++nt)
            zacc[nt] = WMMA_BF16(aP, vb[nt], zacc[nt]);
        __syncthreads();
    }

    // ---- normalize and store z as [B, T, H*Dh] row-major (A-friendly) ----
    bf16* zb = zout + (b * T_ + t0) * (H_ * DH_) + h * DH_;
#pragma unroll
    for (int nt = 0; nt < 4; ++nt) {
        int d = nt * 16 + lm;
#pragma unroll
        for (int r = 0; r < 8; ++r) {
            int m = r + 8 * lg;
            zb[m * (H_ * DH_) + d] = (bf16)(zacc[nt][r] / lrun[r]);
        }
    }
}

// =====================================================================
// Kernel 3: output projection  out[4096,1024] = z[4096,1024] x Wout + b
// Explicitly pipelined like kernel 1.
// =====================================================================
__global__ __launch_bounds__(32) void oproj_kernel(
    const bf16* __restrict__ z, const bf16* __restrict__ wopack,
    const float* __restrict__ bout, float* __restrict__ out) {
    int bid = blockIdx.x;
    int bn = bid & 15;    // 16 column tiles of 64
    int rt = bid >> 4;    // 256 row tiles of 16
    int lane = threadIdx.x, lm = lane & 15, lg = lane >> 4;
    int row0 = rt * 16;

    const bf16* zrow = z + (row0 + lm) * (H_ * DH_);
    v8f acc[4];
#pragma unroll
    for (int i = 0; i < 4; ++i) acc[i] = vzero();

    // prologue
    v8bf a0 = *(const v8bf*)(zrow + lg * 8);
    v8bf a1 = *(const v8bf*)(zrow + 16 + lg * 8);
    v16bf cb[4];
#pragma unroll
    for (int nt = 0; nt < 4; ++nt)
        cb[nt] = *(const v16bf*)(wopack + (bn * 4 + nt) * 512 + lane * 16);

    for (int kc = 0; kc < (H_ * DH_) / 32; ++kc) {
        v16bf a;
#pragma unroll
        for (int i = 0; i < 8; ++i) { a[i] = a0[i]; a[8 + i] = a1[i]; }
        a0 = *(const v8bf*)(zrow + (kc + 1) * 32 + lg * 8);
        a1 = *(const v8bf*)(zrow + (kc + 1) * 32 + 16 + lg * 8);
#pragma unroll
        for (int nt = 0; nt < 4; ++nt) {
            v16bf nb = *(const v16bf*)(wopack +
                ((kc + 1) * (D_ / 16) + bn * 4 + nt) * 512 + lane * 16);
            asm volatile("" : "+v"(acc[nt]) :: "memory");
            acc[nt] = WMMA_BF16(a, cb[nt], acc[nt]);
            cb[nt] = nb;
        }
    }
#pragma unroll
    for (int nt = 0; nt < 4; ++nt) {
        int col = bn * 64 + nt * 16 + lm;
        float bb = bout[col];
#pragma unroll
        for (int r = 0; r < 8; ++r) {
            int m = r + 8 * lg;
            out[(row0 + m) * D_ + col] = acc[nt][r] + bb;
        }
    }
}

// =====================================================================
extern "C" void kernel_launch(void* const* d_in, const int* in_sizes, int n_in,
                              void* d_out, int out_size, void* d_ws, size_t ws_size,
                              hipStream_t stream) {
    const float* res = (const float*)d_in[0];
    const float* WQ  = (const float*)d_in[1];
    const float* WK  = (const float*)d_in[2];
    const float* WV  = (const float*)d_in[3];
    const float* WO  = (const float*)d_in[4];
    const float* bQ  = (const float*)d_in[5];
    const float* bK  = (const float*)d_in[6];
    const float* bV  = (const float*)d_in[7];
    const float* bO  = (const float*)d_in[8];
    float* out = (float*)d_out;

    char* ws = (char*)d_ws;
    const size_t MB = 1024 * 1024;
    bf16* res_bf = (bf16*)(ws + 0 * MB);   // 8 MB   [B,T,D] bf16
    bf16* wq_p   = (bf16*)(ws + 8 * MB);   // 2 MB   packed B-layout
    bf16* wk_p   = (bf16*)(ws + 10 * MB);  // 2 MB
    bf16* wv_p   = (bf16*)(ws + 12 * MB);  // 2 MB
    bf16* wo_p   = (bf16*)(ws + 14 * MB);  // 2 MB
    bf16* qbuf   = (bf16*)(ws + 16 * MB);  // 8 MB   [B,H,T,Dh]
    bf16* kpack  = (bf16*)(ws + 24 * MB);  // 8 MB   packed
    bf16* vpack  = (bf16*)(ws + 32 * MB);  // 8 MB   packed
    bf16* zbuf   = (bf16*)(ws + 40 * MB);  // 8 MB   [B,T,H*Dh]
    // 48 MB used; tail of workspace additionally absorbs pipeline prefetch
    // overrun reads (values never consumed).

    int nres = B_ * T_ * D_;
    cvt_bf16_kernel<<<(nres + 255) / 256, 256, 0, stream>>>(res, res_bf, nres);
    int nw = H_ * D_ * DH_;
    pack_b_kernel<<<(nw + 255) / 256, 256, 0, stream>>>(WQ, wq_p, D_, DH_, H_);
    pack_b_kernel<<<(nw + 255) / 256, 256, 0, stream>>>(WK, wk_p, D_, DH_, H_);
    pack_b_kernel<<<(nw + 255) / 256, 256, 0, stream>>>(WV, wv_p, D_, DH_, H_);
    int nwo = D_ * D_;
    pack_b_kernel<<<(nwo + 255) / 256, 256, 0, stream>>>(WO, wo_p, D_, D_, 1);

    int ntiles = B_ * H_ * (T_ / 16);   // 4096
    qkv_kernel<<<ntiles, 32, 0, stream>>>(res_bf, wq_p, wk_p, wv_p, bQ, bK, bV,
                                          qbuf, kpack, vpack);
    attn_kernel<<<ntiles, 32, 0, stream>>>(qbuf, kpack, vpack, zbuf);
    oproj_kernel<<<(B_ * T_ / 16) * (D_ / 64), 32, 0, stream>>>(zbuf, wo_p, bO, out);
}
